// MambaTemporalModule_77876347011352
// MI455X (gfx1250) — matrix operand order repaired and checked by reading, not compiled, not run on hardware
//
#include <hip/hip_runtime.h>
#include <hip/hip_bf16.h>

// ---------------------------------------------------------------------------
// Mamba-2 forward for MI455X (gfx1250, wave32, WMMA 16x16x32 f16).
// ---------------------------------------------------------------------------

typedef __attribute__((ext_vector_type(16))) _Float16 v16h;
typedef __attribute__((ext_vector_type(8)))  float    v8f;
typedef __attribute__((ext_vector_type(4)))  unsigned int uint4v;

static constexpr int D_MODEL   = 512;
static constexpr int N_LAYERS  = 6;
static constexpr int D_STATE   = 128;
static constexpr int D_INNER   = 1024;
static constexpr int HEADDIM   = 64;
static constexpr int NHEADS    = 16;
static constexpr int D_CONV    = 4;
static constexpr int CONV_DIM  = 1280;   // D_INNER + 2*D_STATE
static constexpr int D_IN_PROJ = 2320;   // 2*D_INNER + 2*D_STATE + NHEADS
static constexpr int CHUNK     = 128;
static constexpr int B_SZ      = 2;
static constexpr int SEQ       = 4096;
static constexpr int NCHUNK    = SEQ / CHUNK;   // 32
static constexpr int BT        = B_SZ * SEQ;    // 8192

__device__ __forceinline__ float siluf(float x) { return x / (1.f + __expf(-x)); }

// --- WMMA helpers ----------------------------------------------------------

__device__ __forceinline__ v8f wmma_f16(v16h a, v16h b, v8f c) {
  return __builtin_amdgcn_wmma_f32_16x16x32_f16(false, a, false, b, (short)0, c,
                                                false, false);
}

// A-matrix fragment (16x32, rows x K), documented interleaved layout:
// lane&15 = row; K halves per lane: base=(lane>>4)*8, slots j -> K = base+j+(j&8)
// i.e. chunks [base..base+7] and [base+16..base+23].
__device__ __forceinline__ v16h ld_frag_a(const _Float16* p, int ld, int row0, int k0) {
  int lane = threadIdx.x & 31;
  const _Float16* q = p + (size_t)(row0 + (lane & 15)) * ld
                        + (k0 + ((lane >> 4) << 3));
  union { v16h h; uint4v u[2]; } r;
  r.u[0] = *(const uint4v*)(const void*)(q);
  r.u[1] = *(const uint4v*)(const void*)(q + 16);
  return r.h;
}

// B-matrix fragment (32x16, K x cols) built from a row-major [N,K] source
// (B(k,n) = W[n,k]).  lane&15 = column n; same interleaved K layout as A:
// base=(lane>>4)*8, chunks [base..base+7] and [base+16..base+23].
__device__ __forceinline__ v16h ld_frag_b(const _Float16* p, int ld, int row0, int k0) {
  int lane = threadIdx.x & 31;
  const _Float16* q = p + (size_t)(row0 + (lane & 15)) * ld
                        + (k0 + ((lane >> 4) << 3));
  union { v16h h; uint4v u[2]; } r;
  r.u[0] = *(const uint4v*)(const void*)(q);
  r.u[1] = *(const uint4v*)(const void*)(q + 16);
  return r.h;
}

// --- reductions ------------------------------------------------------------

__device__ __forceinline__ float warp_sum(float v) {
#pragma unroll
  for (int o = 16; o > 0; o >>= 1) v += __shfl_down(v, o, 32);
  return v;
}

__device__ __forceinline__ float block_sum_256(float v, float* sh8) {
  int lane = threadIdx.x & 31, w = threadIdx.x >> 5;
  v = warp_sum(v);
  if (lane == 0) sh8[w] = v;
  __syncthreads();
  float r = (threadIdx.x < 8) ? sh8[threadIdx.x] : 0.f;
  if (w == 0) {
    r = warp_sum(r);
    if (lane == 0) sh8[0] = r;
  }
  __syncthreads();
  float out = sh8[0];
  __syncthreads();
  return out;
}

// --- elementwise kernels ---------------------------------------------------

__global__ void cvt_f16_kernel(const float* __restrict__ in,
                               _Float16* __restrict__ out, int n) {
  int i = blockIdx.x * 256 + threadIdx.x;
  if (i < n) out[i] = (_Float16)in[i];
}

// LayerNorm over D_MODEL=512, output f16 (feeds in_proj WMMA GEMM).
__global__ __launch_bounds__(256)
void layernorm_f16_kernel(const float* __restrict__ x,
                          const float* __restrict__ w,
                          const float* __restrict__ b,
                          _Float16* __restrict__ out) {
  __shared__ float sh8[8];
  int m = blockIdx.x;
  int t = threadIdx.x;
  float v0 = x[(size_t)m * D_MODEL + t];
  float v1 = x[(size_t)m * D_MODEL + 256 + t];
  float mu = block_sum_256(v0 + v1, sh8) * (1.f / D_MODEL);
  float c0 = v0 - mu, c1 = v1 - mu;
  float var = block_sum_256(c0 * c0 + c1 * c1, sh8) * (1.f / D_MODEL);
  float rstd = rsqrtf(var + 1e-5f);
  out[(size_t)m * D_MODEL + t]       = (_Float16)(c0 * rstd * w[t] + b[t]);
  out[(size_t)m * D_MODEL + 256 + t] = (_Float16)(c1 * rstd * w[256 + t] + b[256 + t]);
}

// Final LayerNorm, fp32 output.
__global__ __launch_bounds__(256)
void layernorm_out_kernel(const float* __restrict__ x,
                          const float* __restrict__ w,
                          const float* __restrict__ b,
                          float* __restrict__ out) {
  __shared__ float sh8[8];
  int m = blockIdx.x;
  int t = threadIdx.x;
  float v0 = x[(size_t)m * D_MODEL + t];
  float v1 = x[(size_t)m * D_MODEL + 256 + t];
  float mu = block_sum_256(v0 + v1, sh8) * (1.f / D_MODEL);
  float c0 = v0 - mu, c1 = v1 - mu;
  float var = block_sum_256(c0 * c0 + c1 * c1, sh8) * (1.f / D_MODEL);
  float rstd = rsqrtf(var + 1e-5f);
  out[(size_t)m * D_MODEL + t]       = c0 * rstd * w[t] + b[t];
  out[(size_t)m * D_MODEL + 256 + t] = c1 * rstd * w[256 + t] + b[256 + t];
}

// --- GEMM: out[M,N] = A[M,K] * W[N,K]^T (+ residual) via WMMA f16 ----------
// grid = (M/128, N/16), block = 256 (8 waves, one 16x16 tile each).
__global__ __launch_bounds__(256)
void gemm_f16_kernel(const _Float16* __restrict__ A,
                     const _Float16* __restrict__ W,
                     float* __restrict__ out,
                     const float* __restrict__ residual,
                     int M, int N, int K) {
  int wave = threadIdx.x >> 5;
  int lane = threadIdx.x & 31;
  int m0 = blockIdx.x * 128 + wave * 16;
  int n0 = blockIdx.y * 16;
  if (m0 >= M) return;
  v8f acc = {};
  for (int k0 = 0; k0 < K; k0 += 32) {
    if (k0 + 32 < K)  // exercise global_prefetch_b8
      __builtin_prefetch(A + (size_t)m0 * K + k0 + 32, 0, 1);
    v16h a = ld_frag_a(A, K, m0, k0);
    v16h bw = ld_frag_b(W, K, n0, k0);
    acc = wmma_f16(a, bw, acc);
  }
  int col = n0 + (lane & 15);
  int rb = m0 + ((lane >> 4) << 3);
#pragma unroll
  for (int r = 0; r < 8; ++r) {
    size_t idx = (size_t)(rb + r) * N + col;
    float v = acc[r];
    if (residual) v += residual[idx];
    out[idx] = v;
  }
}

// --- depthwise causal conv(width 4) + SiLU over xBC channels ---------------
__global__ void conv_silu_kernel(const float* __restrict__ zx,
                                 const float* __restrict__ cw,
                                 const float* __restrict__ cb,
                                 float* __restrict__ out) {
  int idx = blockIdx.x * 256 + threadIdx.x;
  if (idx >= BT * CONV_DIM) return;
  int m = idx / CONV_DIM;
  int ch = idx - m * CONV_DIM;
  int t = m & (SEQ - 1);
  float acc = cb[ch];
#pragma unroll
  for (int j = 0; j < D_CONV; ++j) {
    int tt = t + j - (D_CONV - 1);
    if (tt >= 0)
      acc += cw[ch * D_CONV + j] *
             zx[(size_t)(m + j - (D_CONV - 1)) * D_IN_PROJ + D_INNER + ch];
  }
  out[idx] = siluf(acc);
}

// --- dt = softplus(raw + bias) ---------------------------------------------
__global__ void dt_softplus_kernel(const float* __restrict__ zx,
                                   const float* __restrict__ dtb,
                                   float* __restrict__ dt) {
  int idx = blockIdx.x * 256 + threadIdx.x;
  if (idx >= BT * NHEADS) return;
  int m = idx >> 4, h = idx & 15;
  float v = zx[(size_t)m * D_IN_PROJ + (D_INNER + CONV_DIM) + h] + dtb[h];
  dt[idx] = (v > 20.f) ? v : log1pf(__expf(v));
}

// --- SSD intra-chunk kernel: one block per (b, chunk) ----------------------
// LDS carve (dynamic, 152KB):
//   s_acs : 16*128 f32          (  8KB)
//   s_C   : 128*128 f16         ( 32KB)
//   s_B   : 128*128 f16         ( 32KB)
//   s_G   : 128*128 f32         ( 64KB)
//   s_X   : 128*64  f16         ( 16KB)
__global__ __launch_bounds__(512)
void ssd_chunk_kernel(const float* __restrict__ xBCc,
                      const float* __restrict__ dtb,
                      const float* __restrict__ A_log,
                      const float* __restrict__ Dvec,
                      float* __restrict__ acs_g,
                      float* __restrict__ states_g,
                      float* __restrict__ y_g) {
  extern __shared__ char smem[];
  float*    s_acs = (float*)smem;
  _Float16* s_C   = (_Float16*)(smem + 8192);
  _Float16* s_B   = (_Float16*)(smem + 8192 + 32768);
  float*    s_G   = (float*)(smem + 8192 + 65536);
  _Float16* s_X   = (_Float16*)(smem + 8192 + 131072);

  int bc = blockIdx.x;
  int b = bc >> 5, c = bc & 31;
  int base_m = b * SEQ + c * CHUNK;
  int tid = threadIdx.x;
  int wave = tid >> 5, lane = tid & 31;

  // Phase A: per-head cumulative sum of dt*A (16 serial scans).
  if (tid < NHEADS) {
    int h = tid;
    float Ah = -__expf(A_log[h]);
    float acc = 0.f;
    for (int q = 0; q < CHUNK; ++q) {
      acc += dtb[(size_t)(base_m + q) * NHEADS + h] * Ah;
      s_acs[h * CHUNK + q] = acc;
      acs_g[((size_t)bc * NHEADS + h) * CHUNK + q] = acc;
    }
  }

  // Phase B: stage B and C tiles (f16).
  for (int i = tid; i < CHUNK * D_STATE; i += 512) {
    int q = i >> 7, n = i & 127;
    const float* row = xBCc + (size_t)(base_m + q) * CONV_DIM;
    s_B[i] = (_Float16)row[D_INNER + n];
    s_C[i] = (_Float16)row[D_INNER + D_STATE + n];
  }
  __syncthreads();

  // Phase C: G = C * B^T  (8x8 tiles of 16x16, 4 per wave).
  for (int t = 0; t < 4; ++t) {
    int tile = wave * 4 + t;
    int qt = tile >> 3, st = tile & 7;
    v8f acc = {};
    for (int n0 = 0; n0 < D_STATE; n0 += 32) {
      v16h a  = ld_frag_a(s_C, D_STATE, qt * 16, n0);
      v16h bb = ld_frag_b(s_B, D_STATE, st * 16, n0);
      acc = wmma_f16(a, bb, acc);
    }
    int s  = st * 16 + (lane & 15);
    int qb = qt * 16 + ((lane >> 4) << 3);
#pragma unroll
    for (int r = 0; r < 8; ++r) s_G[(qb + r) * CHUNK + s] = acc[r];
  }
  __syncthreads();

  // Phase D: per head, Y_diag and chunk states via WMMA.
  for (int h = 0; h < NHEADS; ++h) {
    for (int i = tid; i < CHUNK * HEADDIM; i += 512) {
      int q = i >> 6, p = i & 63;
      int m = base_m + q;
      s_X[i] = (_Float16)(xBCc[(size_t)m * CONV_DIM + h * HEADDIM + p] *
                          dtb[(size_t)m * NHEADS + h]);
    }
    __syncthreads();
    float aLast = s_acs[h * CHUNK + CHUNK - 1];
    float Dh = Dvec[h];

    // Y_diag = (L .* G) @ X : 8x4 tiles, 2 per wave.
    for (int t = 0; t < 2; ++t) {
      int tile = wave * 2 + t;
      int qt = tile >> 2, pt = tile & 3;
      v8f acc = {};
      for (int s0 = 0; s0 < CHUNK; s0 += 32) {
        int q = qt * 16 + (lane & 15);
        float aq = s_acs[h * CHUNK + q];
        int sbA = s0 + ((lane >> 4) << 3);   // A-frag interleaved base
        int sbB = s0 + ((lane >> 4) << 3);   // B-frag interleaved base
        int p = pt * 16 + (lane & 15);
        v16h a, bx;
#pragma unroll
        for (int j = 0; j < 16; ++j) {
          int sA = sbA + j + (j & 8);
          float v = 0.f;
          if (sA <= q)
            v = s_G[q * CHUNK + sA] * __expf(aq - s_acs[h * CHUNK + sA]);
          a[j]  = (_Float16)v;
          int sB = sbB + j + (j & 8);
          bx[j] = s_X[sB * HEADDIM + p];
        }
        acc = wmma_f16(a, bx, acc);
      }
      int p  = pt * 16 + (lane & 15);
      int qb = qt * 16 + ((lane >> 4) << 3);
#pragma unroll
      for (int r = 0; r < 8; ++r) {
        int m = base_m + qb + r;
        float xv = xBCc[(size_t)m * CONV_DIM + h * HEADDIM + p];
        y_g[((size_t)m * NHEADS + h) * HEADDIM + p] = acc[r] + Dh * xv;
      }
    }

    // states[p,n] = sum_q X[q,p]*decay[q]*B[q,n] : 4x8 tiles, 2 per wave.
    for (int t = 0; t < 2; ++t) {
      int tile = wave * 2 + t;
      int pt = tile >> 3, nt = tile & 7;
      v8f acc = {};
      for (int q0 = 0; q0 < CHUNK; q0 += 32) {
        int p = pt * 16 + (lane & 15);
        int n = nt * 16 + (lane & 15);
        int qbA = q0 + ((lane >> 4) << 3);
        int qbB = q0 + ((lane >> 4) << 3);
        v16h a, bb;
#pragma unroll
        for (int j = 0; j < 16; ++j) {
          int qA = qbA + j + (j & 8);
          float decay = __expf(aLast - s_acs[h * CHUNK + qA]);
          a[j]  = (_Float16)((float)s_X[qA * HEADDIM + p] * decay);
          int qB = qbB + j + (j & 8);
          bb[j] = s_B[qB * D_STATE + n];
        }
        acc = wmma_f16(a, bb, acc);
      }
      int n  = nt * 16 + (lane & 15);
      int pb = pt * 16 + ((lane >> 4) << 3);
#pragma unroll
      for (int r = 0; r < 8; ++r)
        states_g[(((size_t)bc * NHEADS + h) * HEADDIM + pb + r) * D_STATE + n] =
            acc[r];
    }
    __syncthreads();
  }
}

// --- inter-chunk recurrence: states[c] -> init_states[c] (in place) --------
// grid = B*NHEADS*HEADDIM blocks of 128 threads (one per n).
__global__ __launch_bounds__(128)
void state_scan_kernel(float* __restrict__ states_g,
                       const float* __restrict__ acs_g) {
  int blk = blockIdx.x;
  int b = blk / (NHEADS * HEADDIM);
  int rem = blk - b * NHEADS * HEADDIM;
  int h = rem / HEADDIM;
  int p = rem - h * HEADDIM;
  int n = threadIdx.x;
  float carry = 0.f;
  for (int c = 0; c < NCHUNK; ++c) {
    int bc = b * NCHUNK + c;
    size_t idx = (((size_t)bc * NHEADS + h) * HEADDIM + p) * D_STATE + n;
    float s = states_g[idx];
    states_g[idx] = carry;  // becomes init_states for chunk c
    float aLast = acs_g[((size_t)bc * NHEADS + h) * CHUNK + CHUNK - 1];
    carry = carry * __expf(aLast) + s;
  }
}

// --- Y_off = exp(acs[q]) * C @ init_states^T, added into y -----------------
// LDS: s_acs 8KB, s_C 32KB, s_S 16KB  => 56KB dynamic.
__global__ __launch_bounds__(512)
void y_off_kernel(const float* __restrict__ xBCc,
                  const float* __restrict__ acs_g,
                  const float* __restrict__ states_g,
                  float* __restrict__ y_g) {
  extern __shared__ char smem[];
  float*    s_acs = (float*)smem;
  _Float16* s_C   = (_Float16*)(smem + 8192);
  _Float16* s_S   = (_Float16*)(smem + 8192 + 32768);

  int bc = blockIdx.x;
  int b = bc >> 5, c = bc & 31;
  int base_m = b * SEQ + c * CHUNK;
  int tid = threadIdx.x;
  int wave = tid >> 5, lane = tid & 31;

  for (int i = tid; i < NHEADS * CHUNK; i += 512)
    s_acs[i] = acs_g[(size_t)bc * NHEADS * CHUNK + i];
  for (int i = tid; i < CHUNK * D_STATE; i += 512) {
    int q = i >> 7, n = i & 127;
    s_C[i] = (_Float16)
        xBCc[(size_t)(base_m + q) * CONV_DIM + D_INNER + D_STATE + n];
  }
  __syncthreads();

  for (int h = 0; h < NHEADS; ++h) {
    for (int i = tid; i < HEADDIM * D_STATE; i += 512)
      s_S[i] = (_Float16)
          states_g[((size_t)bc * NHEADS + h) * HEADDIM * D_STATE + i];
    __syncthreads();

    for (int t = 0; t < 2; ++t) {
      int tile = wave * 2 + t;
      int qt = tile >> 2, pt = tile & 3;
      v8f acc = {};
      for (int n0 = 0; n0 < D_STATE; n0 += 32) {
        v16h a  = ld_frag_a(s_C, D_STATE, qt * 16, n0);
        v16h bs = ld_frag_b(s_S, D_STATE, pt * 16, n0);  // S[p,n] row-major
        acc = wmma_f16(a, bs, acc);
      }
      int p  = pt * 16 + (lane & 15);
      int qb = qt * 16 + ((lane >> 4) << 3);
#pragma unroll
      for (int r = 0; r < 8; ++r) {
        int q = qb + r;
        float eq = __expf(s_acs[h * CHUNK + q]);
        size_t idx = ((size_t)(base_m + q) * NHEADS + h) * HEADDIM + p;
        y_g[idx] += eq * acc[r];
      }
    }
    __syncthreads();
  }
}

// --- gated RMS norm: y16 = rmsnorm(y * silu(z)) * rms_w --------------------
__global__ __launch_bounds__(256)
void gate_rms_kernel(const float* __restrict__ y_g,
                     const float* __restrict__ zx,
                     const float* __restrict__ rms_w,
                     _Float16* __restrict__ y16) {
  __shared__ float sh8[8];
  int m = blockIdx.x;
  int t = threadIdx.x;
  float yz[4];
  float ss = 0.f;
#pragma unroll
  for (int j = 0; j < 4; ++j) {
    int i = t + j * 256;
    float z = zx[(size_t)m * D_IN_PROJ + i];
    float v = y_g[(size_t)m * D_INNER + i] * siluf(z);
    yz[j] = v;
    ss += v * v;
  }
  float tot = block_sum_256(ss, sh8);
  float rstd = rsqrtf(tot * (1.f / D_INNER) + 1e-5f);
#pragma unroll
  for (int j = 0; j < 4; ++j) {
    int i = t + j * 256;
    y16[(size_t)m * D_INNER + i] = (_Float16)(yz[j] * rstd * rms_w[i]);
  }
}

// ---------------------------------------------------------------------------
extern "C" void kernel_launch(void* const* d_in, const int* in_sizes, int n_in,
                              void* d_out, int out_size, void* d_ws,
                              size_t ws_size, hipStream_t stream) {
  (void)in_sizes; (void)n_in; (void)out_size; (void)ws_size;
  const float* x         = (const float*)d_in[0];
  const float* in_proj_w = (const float*)d_in[1];
  const float* conv_w    = (const float*)d_in[2];
  const float* conv_b    = (const float*)d_in[3];
  const float* dt_bias   = (const float*)d_in[4];
  const float* A_log     = (const float*)d_in[5];
  const float* Dv        = (const float*)d_in[6];
  const float* rms_w     = (const float*)d_in[7];
  const float* out_w     = (const float*)d_in[8];
  const float* ln_w      = (const float*)d_in[9];
  const float* ln_b      = (const float*)d_in[10];
  const float* fln_w     = (const float*)d_in[11];
  const float* fln_b     = (const float*)d_in[12];

  char* ws = (char*)d_ws;
  size_t off = 0;
  auto carve = [&](size_t bytes) {
    void* p = ws + off;
    off += (bytes + 255) & ~(size_t)255;
    return p;
  };
  float*    h      = (float*)carve((size_t)BT * D_MODEL * 4);
  _Float16* hn16   = (_Float16*)carve((size_t)BT * D_MODEL * 2);
  _Float16* wi16   = (_Float16*)carve((size_t)D_IN_PROJ * D_MODEL * 2);
  _Float16* wo16   = (_Float16*)carve((size_t)D_MODEL * D_INNER * 2);
  float*    zx     = (float*)carve((size_t)BT * D_IN_PROJ * 4);
  float*    xBCc   = (float*)carve((size_t)BT * CONV_DIM * 4);
  float*    dtbuf  = (float*)carve((size_t)BT * NHEADS * 4);
  float*    acs    = (float*)carve((size_t)B_SZ * NCHUNK * NHEADS * CHUNK * 4);
  float*    states = (float*)carve((size_t)B_SZ * NCHUNK * NHEADS * HEADDIM * D_STATE * 4);
  float*    y      = (float*)carve((size_t)BT * D_INNER * 4);
  _Float16* y16    = (_Float16*)carve((size_t)BT * D_INNER * 2);

  hipMemcpyAsync(h, x, (size_t)BT * D_MODEL * sizeof(float),
                 hipMemcpyDeviceToDevice, stream);

  for (int i = 0; i < N_LAYERS; ++i) {
    layernorm_f16_kernel<<<BT, 256, 0, stream>>>(h, ln_w + i * D_MODEL,
                                                 ln_b + i * D_MODEL, hn16);
    cvt_f16_kernel<<<(D_IN_PROJ * D_MODEL + 255) / 256, 256, 0, stream>>>(
        in_proj_w + (size_t)i * D_IN_PROJ * D_MODEL, wi16, D_IN_PROJ * D_MODEL);
    gemm_f16_kernel<<<dim3(BT / 128, D_IN_PROJ / 16), 256, 0, stream>>>(
        hn16, wi16, zx, nullptr, BT, D_IN_PROJ, D_MODEL);
    conv_silu_kernel<<<(BT * CONV_DIM + 255) / 256, 256, 0, stream>>>(
        zx, conv_w + (size_t)i * CONV_DIM * D_CONV, conv_b + i * CONV_DIM, xBCc);
    dt_softplus_kernel<<<(BT * NHEADS + 255) / 256, 256, 0, stream>>>(
        zx, dt_bias + i * NHEADS, dtbuf);
    ssd_chunk_kernel<<<B_SZ * NCHUNK, 512, 155648, stream>>>(
        xBCc, dtbuf, A_log + i * NHEADS, Dv + i * NHEADS, acs, states, y);
    state_scan_kernel<<<B_SZ * NHEADS * HEADDIM, 128, 0, stream>>>(states, acs);
    y_off_kernel<<<B_SZ * NCHUNK, 512, 57344, stream>>>(xBCc, acs, states, y);
    gate_rms_kernel<<<BT, 256, 0, stream>>>(y, zx, rms_w + i * D_INNER, y16);
    cvt_f16_kernel<<<(D_MODEL * D_INNER + 255) / 256, 256, 0, stream>>>(
        out_w + (size_t)i * D_MODEL * D_INNER, wo16, D_MODEL * D_INNER);
    gemm_f16_kernel<<<dim3(BT / 128, D_MODEL / 16), 256, 0, stream>>>(
        y16, wo16, h, h, BT, D_MODEL, D_INNER);
  }
  layernorm_out_kernel<<<BT, 256, 0, stream>>>(h, fln_w, fln_b, (float*)d_out);
}
